// EncoderLayer_70540542869696
// MI455X (gfx1250) — compile-verified, matrix operand order checked by reading, not run on hardware
//
#include <hip/hip_runtime.h>

// ---------------- types / helpers ----------------
typedef __attribute__((ext_vector_type(16))) __bf16 v16bf;
typedef __attribute__((ext_vector_type(8)))  float  v8f;
typedef __attribute__((ext_vector_type(4)))  unsigned int uint4v;
typedef __attribute__((ext_vector_type(4)))  unsigned int v4u;
typedef __attribute__((ext_vector_type(8)))  int v8i_t;
typedef __attribute__((ext_vector_type(4)))  int v4i_t;

#if __has_builtin(__builtin_amdgcn_tensor_load_to_lds)
#define HAVE_TDM 1
#else
#define HAVE_TDM 0
#endif

__device__ __forceinline__ unsigned short f2bf(float f) {
  union { float f; unsigned int u; } a; a.f = f;
  unsigned int u = a.u;
  u += 0x7fffu + ((u >> 16) & 1u);           // round-to-nearest-even
  return (unsigned short)(u >> 16);
}

// One 16xbf16 WMMA operand fragment. For both A (16x32, row M per lane) and
// B (32x16, col N per lane, staged K-contiguous) the per-lane data is
// 8 contiguous bf16 at p[0..7] (K=koff+0..7) and 8 at p[16..23]
// (K=koff+16..23), koff=(lane<16)?0:8.  p is 16B-aligned by construction.
__device__ __forceinline__ v16bf frag16(const unsigned short* p) {
  union { v16bf v; uint4v q[2]; } f;
  f.q[0] = *(const uint4v*)p;
  f.q[1] = *(const uint4v*)(p + 16);
  return f.v;
}

__device__ __forceinline__ v8f zero8() {
  v8f z = {0.f, 0.f, 0.f, 0.f, 0.f, 0.f, 0.f, 0.f};
  return z;
}

__device__ __forceinline__ v8f wmma_bf16(v16bf a, v16bf b, v8f c) {
  return __builtin_amdgcn_wmma_f32_16x16x32_bf16(false, a, false, b,
                                                 (short)0, c, false, false);
}

#if HAVE_TDM
// Issue one TDM 2D tile load (16-bit elements) global -> LDS.
// D# encoding per CDNA5 ISA ch.8: group0 = {count, lds_addr, global_addr,
// type=2}; group1 = {data_size=2B, tensor dims (huge: no OOB clip),
// tile_dim0/1, tensor_dim0_stride}.  tile rows land packed in LDS.
// All operands are wave-uniform scalars. Tracked by TENSORcnt.
// This toolchain exposes the 6-arg builtin form:
//   (uint32x4 g0, int32x8 g1, int32x4, int32x4, int32x8, i32 cpol)
__device__ __forceinline__ void tdm_load_2d(unsigned lds_off, const void* gptr,
                                            unsigned tile_w, unsigned tile_h,
                                            unsigned long long row_stride) {
  const unsigned long long ga = (unsigned long long)gptr;
  v4u g0;
  g0.x = 1u;                                                  // count=1, user D#
  g0.y = lds_off;                                             // lds_addr bytes
  g0.z = (unsigned)ga;                                        // global_addr lo
  g0.w = (unsigned)((ga >> 32) & 0x01FFFFFFu) | (2u << 30);   // hi | type=2
  v8i_t g1;
  g1[0] = (int)(1u << 16);                         // wg_mask=0, data_size=1(2B)
  g1[1] = (int)0xFFFF0000u;                        // tensor_dim0[15:0]
  g1[2] = (int)0xFFFF7FFFu;                        // td0[31:16] | td1[15:0]
  g1[3] = (int)(0x7FFFu | (tile_w << 16));         // td1[31:16] | tile_dim0
  g1[4] = (int)(tile_h & 0xFFFFu);                 // tile_dim1 (tile_dim2=0)
  g1[5] = (int)(unsigned)(row_stride & 0xFFFFFFFFull);     // dim0_stride lo
  g1[6] = (int)(unsigned)((row_stride >> 32) & 0xFFFFull); // dim0_stride hi
  g1[7] = 0;
  const v4i_t gz4 = {0, 0, 0, 0};
  const v8i_t gz8 = {0, 0, 0, 0, 0, 0, 0, 0};
  __builtin_amdgcn_tensor_load_to_lds(g0, g1, gz4, gz4, gz8, 0);
}
#endif

// ---------------- one-time weight prep: f32 [K][N] -> bf16 [N][K] ----------
__global__ __launch_bounds__(256)
void wprep_kernel(const float* __restrict__ in, unsigned short* __restrict__ out,
                  int K, int N) {
  const int idx = blockIdx.x * 256 + threadIdx.x;  // over N*K, out-coalesced
  const int n = idx / K;
  const int k = idx - n * K;
  out[idx] = f2bf(in[(size_t)k * N + n]);
}

// ---------------- LayerNorm (f32 in -> bf16 out) ----------------
__global__ __launch_bounds__(256)
void ln_bf16_kernel(const float* __restrict__ x, const float* __restrict__ g,
                    const float* __restrict__ b, unsigned short* __restrict__ out) {
  __shared__ float red[8];
  const int row = blockIdx.x;
  const int t   = threadIdx.x;               // 256 threads = 8 waves
  const float v = x[(size_t)row * 256 + t];

  float s = v;
#pragma unroll
  for (int m = 16; m >= 1; m >>= 1) s += __shfl_xor(s, m, 32);
  if ((t & 31) == 0) red[t >> 5] = s;
  __syncthreads();
  float tot = 0.f;
#pragma unroll
  for (int i = 0; i < 8; ++i) tot += red[i];
  const float mu = tot * (1.0f / 256.0f);
  const float dv = v - mu;
  __syncthreads();

  float s2 = dv * dv;
#pragma unroll
  for (int m = 16; m >= 1; m >>= 1) s2 += __shfl_xor(s2, m, 32);
  if ((t & 31) == 0) red[t >> 5] = s2;
  __syncthreads();
  float tot2 = 0.f;
#pragma unroll
  for (int i = 0; i < 8; ++i) tot2 += red[i];
  const float rstd = __frsqrt_rn(tot2 * (1.0f / 256.0f) + 1e-5f);

  out[(size_t)row * 256 + t] = f2bf(dv * rstd * g[t] + b[t]);
}

// ---------------- bf16 WMMA GEMM, 128x128 block tile ----------------
// C[M,N] = A[M,K](bf16) * Bt[N][K](bf16, pre-transposed) (+bias)(+resid)(relu)
// A addressing supports head-chunked K:
//   addr = A + (k>>chunk_shift)*chunk_stride + row*a_row_stride + (k&chunk_mask)
// 256 threads = 8 waves in a 4(M)x2(N) grid; wave tile 32x64
// (2 A-frags x 4 B-frags = 8 WMMAs per K-step of 32).
// Double-buffered LDS, one barrier per K-step.  Tiles are staged with the
// Tensor Data Mover (tensor_load_to_lds, TENSORcnt) issued by wave 0; the
// next buffer's DMA runs while all waves execute the WMMA burst.
template <bool BIAS, bool RESID, bool RELU, bool OUTF_EN, bool OUTB_EN>
__global__ __launch_bounds__(256)
void gemm_bf16_kernel(const unsigned short* __restrict__ A, int a_row_stride,
                      int a_chunk_shift, size_t a_chunk_stride,
                      const unsigned short* __restrict__ Bt, int b_row_stride,
                      const float* __restrict__ bias,
                      const float* __restrict__ resid,
                      float* __restrict__ outF, unsigned short* __restrict__ outB,
                      int N, int K) {
  __shared__ unsigned short As[2][128 * 32];   // [m][k], stride 32
  __shared__ unsigned short Bs[2][128 * 32];   // [n][k], stride 32

  const int m0 = blockIdx.x * 128;
  const int n0 = blockIdx.y * 128;
  const int wave  = threadIdx.x >> 5;
  const int lane  = threadIdx.x & 31;
  const int lan15 = lane & 15;
  const int koff  = (lane < 16) ? 0 : 8;
  const int wy = wave >> 1;                    // 0..3  (M groups of 32)
  const int wx = wave & 1;                     // 0..1  (N groups of 64)
  const int a_chunk_mask = (1 << a_chunk_shift) - 1;

  v8f acc[2][4];
#pragma unroll
  for (int i = 0; i < 2; ++i)
#pragma unroll
    for (int j = 0; j < 4; ++j) acc[i][j] = zero8();

#if HAVE_TDM
  const unsigned ldsA[2] = {(unsigned)(size_t)&As[0][0],
                            (unsigned)(size_t)&As[1][0]};
  const unsigned ldsB[2] = {(unsigned)(size_t)&Bs[0][0],
                            (unsigned)(size_t)&Bs[1][0]};
  auto issue_tiles = [&](int k0, int buf) {
    const int cidx = k0 >> a_chunk_shift;
    tdm_load_2d(ldsA[buf],
                A + (size_t)cidx * a_chunk_stride +
                    (size_t)m0 * a_row_stride + (k0 & a_chunk_mask),
                32, 128, (unsigned long long)a_row_stride);
    tdm_load_2d(ldsB[buf], Bt + (size_t)n0 * b_row_stride + k0,
                32, 128, (unsigned long long)b_row_stride);
  };
  if (wave == 0) issue_tiles(0, 0);
#else
  // fallback: manual staging through VGPRs
  uint4v aReg[2], bReg[2];
  int r_[2], c_[2];
#pragma unroll
  for (int c = 0; c < 2; ++c) {
    const int idx = (threadIdx.x + c * 256) * 8;
    r_[c] = idx >> 5;
    c_[c] = idx & 31;
  }
  auto stage_global = [&](int k0) {
#pragma unroll
    for (int c = 0; c < 2; ++c) {
      const int gk = k0 + c_[c];
      const int cidx = gk >> a_chunk_shift;
      aReg[c] = *(const uint4v*)(A + (size_t)cidx * a_chunk_stride +
                                 (size_t)(m0 + r_[c]) * a_row_stride +
                                 (gk & a_chunk_mask));
      bReg[c] = *(const uint4v*)(Bt + (size_t)(n0 + r_[c]) * b_row_stride + gk);
    }
  };
  auto store_lds = [&](int buf) {
#pragma unroll
    for (int c = 0; c < 2; ++c) {
      *(uint4v*)&As[buf][r_[c] * 32 + c_[c]] = aReg[c];
      *(uint4v*)&Bs[buf][r_[c] * 32 + c_[c]] = bReg[c];
    }
  };
  stage_global(0);
#endif

  int cur = 0;
  for (int k0 = 0; k0 < K; k0 += 32) {
#if HAVE_TDM
    if (wave == 0) __builtin_amdgcn_s_wait_tensorcnt(0);  // buf cur landed
    __syncthreads();
    if (wave == 0 && k0 + 32 < K) issue_tiles(k0 + 32, cur ^ 1);
#else
    store_lds(cur);
    __syncthreads();
    if (k0 + 32 < K) stage_global(k0 + 32);
#endif

    const unsigned short* as = As[cur];
    const unsigned short* bs = Bs[cur];
    const v16bf af0 = frag16(&as[(wy * 32 + lan15) * 32 + koff]);
    const v16bf af1 = frag16(&as[(wy * 32 + 16 + lan15) * 32 + koff]);
#pragma unroll
    for (int nt = 0; nt < 4; ++nt) {
      const v16bf bfr = frag16(&bs[(wx * 64 + nt * 16 + lan15) * 32 + koff]);
      acc[0][nt] = wmma_bf16(af0, bfr, acc[0][nt]);
      acc[1][nt] = wmma_bf16(af1, bfr, acc[1][nt]);
    }
    cur ^= 1;
  }

  // branch-free epilogue (C layout: lane holds col lan15; VGPR g = row g/g+8)
#pragma unroll
  for (int mt = 0; mt < 2; ++mt) {
    const int rbase = m0 + wy * 32 + mt * 16 + ((lane < 16) ? 0 : 8);
#pragma unroll
    for (int nt = 0; nt < 4; ++nt) {
      const int col = n0 + wx * 64 + nt * 16 + lan15;
      float bv = 0.f;
      if constexpr (BIAS) bv = bias[col];
#pragma unroll
      for (int g = 0; g < 8; ++g) {
        const size_t r = (size_t)(rbase + g);
        float v = acc[mt][nt][g] + bv;
        if constexpr (RESID) v += resid[r * N + col];
        if constexpr (RELU)  v = fmaxf(v, 0.0f);
        if constexpr (OUTF_EN) outF[r * N + col] = v;
        if constexpr (OUTB_EN) outB[r * N + col] = f2bf(v);
      }
    }
  }
}

// ---------------- Flash attention ----------------
// qkv: bf16 [12][BS][256] laid out as (t*4+h) with t=0:Q 1:K 2:V,
//      BS row index = b*2048 + s.  head dim = 256, scale = 1/16.
// Block: 128 threads (4 waves); 64 query rows per block, 64-key tiles.
// K tiles are staged with the TDM; V needs a transpose so it stays manual.
#define ATT_S  2048
#define ATT_BS 16384
#define ATT_D  256

__global__ __launch_bounds__(128)
void attn_kernel(const unsigned short* __restrict__ qkv,
                 unsigned short* __restrict__ attn_out) {
  __shared__ unsigned short KV[64 * 256];      // K tile [kr][e] then V^T [e][kr]
  __shared__ unsigned short Ps[4][16 * 64];    // per-wave probabilities (bf16)

  const int qblk = blockIdx.x;                 // 0..31
  const int bh   = blockIdx.y;                 // 0..31
  const int b    = bh >> 2;
  const int h    = bh & 3;
  const size_t base_row = (size_t)b * ATT_S;

  const unsigned short* Q  = qkv + (size_t)(0 + h) * ATT_BS * ATT_D;
  const unsigned short* Kp = qkv + (size_t)(4 + h) * ATT_BS * ATT_D;
  const unsigned short* Vp = qkv + (size_t)(8 + h) * ATT_BS * ATT_D;

  const int wave = threadIdx.x >> 5;
  const int lane = threadIdx.x & 31;
  const int lan15 = lane & 15;
  const int koff  = (lane < 16) ? 0 : 8;
  const int q0 = qblk * 64;
#if HAVE_TDM
  const unsigned ldsKV = (unsigned)(size_t)&KV[0];
#endif

  // Q fragments for this wave's 16 rows, all 8 K-steps
  v16bf qf[8];
  {
    const unsigned short* qr = Q + (base_row + q0 + wave * 16 + lan15) * ATT_D;
#pragma unroll
    for (int ks = 0; ks < 8; ++ks) qf[ks] = frag16(qr + ks * 32 + koff);
  }

  float mrow[8], lrow[8];
  v8f oacc[16];
#pragma unroll
  for (int g = 0; g < 8; ++g) { mrow[g] = -1e30f; lrow[g] = 0.f; }
#pragma unroll
  for (int i = 0; i < 16; ++i) oacc[i] = zero8();

  for (int kt = 0; kt < ATT_S / 64; ++kt) {
    __syncthreads();                          // prev PV done with KV / Ps
    // stage K tile row-major [kr][e]
#if HAVE_TDM
    if (wave == 0) {
      tdm_load_2d(ldsKV, Kp + (base_row + kt * 64) * ATT_D, 256, 64, 256ull);
      __builtin_amdgcn_s_wait_tensorcnt(0);
    }
#else
#pragma unroll
    for (int c = 0; c < 16; ++c) {
      const int idx = (threadIdx.x + c * 128) * 8;
      const int kr = idx >> 8, e = idx & 255;
      *(uint4v*)&KV[kr * 256 + e] =
          *(const uint4v*)(Kp + (base_row + kt * 64 + kr) * ATT_D + e);
    }
#endif
    __syncthreads();

    // scores = Q * K^T * scale  (4 tiles of 16 key columns per wave)
    v8f sc[4];
#pragma unroll
    for (int nt = 0; nt < 4; ++nt) {
      sc[nt] = zero8();
#pragma unroll
      for (int ks = 0; ks < 8; ++ks) {
        const v16bf bfr = frag16(&KV[(nt * 16 + lan15) * 256 + ks * 32 + koff]);
        sc[nt] = wmma_bf16(qf[ks], bfr, sc[nt]);
      }
#pragma unroll
      for (int g = 0; g < 8; ++g) sc[nt][g] *= 0.0625f;   // 1/sqrt(256)
    }

    // online softmax (row reductions across the 16-lane half via shfl_xor)
    float sf[8], rsum[8];
#pragma unroll
    for (int g = 0; g < 8; ++g) {
      float m = fmaxf(fmaxf(sc[0][g], sc[1][g]), fmaxf(sc[2][g], sc[3][g]));
#pragma unroll
      for (int msk = 8; msk >= 1; msk >>= 1) m = fmaxf(m, __shfl_xor(m, msk, 32));
      const float mnew = fmaxf(mrow[g], m);
      sf[g] = __expf(mrow[g] - mnew);
      mrow[g] = mnew;
      float rs = 0.f;
#pragma unroll
      for (int nt = 0; nt < 4; ++nt) {
        const float p = __expf(sc[nt][g] - mnew);
        sc[nt][g] = p;
        rs += p;
      }
#pragma unroll
      for (int msk = 8; msk >= 1; msk >>= 1) rs += __shfl_xor(rs, msk, 32);
      rsum[g] = rs;
    }
#pragma unroll
    for (int g = 0; g < 8; ++g) lrow[g] = lrow[g] * sf[g] + rsum[g];
#pragma unroll
    for (int i = 0; i < 16; ++i)
#pragma unroll
      for (int g = 0; g < 8; ++g) oacc[i][g] *= sf[g];

    // C-layout -> A-layout via per-wave LDS round trip (bf16 probs)
    {
      unsigned short* pw = Ps[wave];
      const int prb = (lane < 16) ? 0 : 8;
#pragma unroll
      for (int nt = 0; nt < 4; ++nt)
#pragma unroll
        for (int g = 0; g < 8; ++g)
          pw[(prb + g) * 64 + nt * 16 + lan15] = f2bf(sc[nt][g]);
    }
    __syncthreads();                          // everyone done reading K rows

    // overwrite KV with V^T: KV[e][kr]
#pragma unroll
    for (int c = 0; c < 16; ++c) {
      const int idx = (threadIdx.x + c * 128) * 8;
      const int r = idx >> 8, e0 = idx & 255;
      union { uint4v q; unsigned short u[8]; } vv;
      vv.q = *(const uint4v*)(Vp + (base_row + kt * 64 + r) * ATT_D + e0);
#pragma unroll
      for (int j = 0; j < 8; ++j) KV[(e0 + j) * 64 + r] = vv.u[j];
    }
    __syncthreads();

    // out += P * V   (A = P[16,64] bf16, B = V[64,256] via V^T tile)
    const unsigned short* pa = Ps[wave];
#pragma unroll
    for (int ksz = 0; ksz < 2; ++ksz) {
      const v16bf af = frag16(&pa[lan15 * 64 + ksz * 32 + koff]);
#pragma unroll
      for (int nt2 = 0; nt2 < 16; ++nt2) {
        const v16bf bfr = frag16(&KV[(nt2 * 16 + lan15) * 64 + ksz * 32 + koff]);
        oacc[nt2] = wmma_bf16(af, bfr, oacc[nt2]);
      }
    }
  }

  // normalize and store (bf16, head-chunked layout [h][BS][256])
  unsigned short* outp = attn_out + (size_t)h * ATT_BS * ATT_D;
  const int rb = q0 + wave * 16 + ((lane < 16) ? 0 : 8);
#pragma unroll
  for (int g = 0; g < 8; ++g) {
    const float inv = 1.0f / lrow[g];
    const size_t r = base_row + rb + g;
#pragma unroll
    for (int nt2 = 0; nt2 < 16; ++nt2)
      outp[r * ATT_D + nt2 * 16 + lan15] = f2bf(oacc[nt2][g] * inv);
  }
}

// ---------------- host launcher ----------------
extern "C" void kernel_launch(void* const* d_in, const int* in_sizes, int n_in,
                              void* d_out, int out_size, void* d_ws, size_t ws_size,
                              hipStream_t stream) {
  const float* x   = (const float*)d_in[0];
  const float* Wq  = (const float*)d_in[1];
  const float* bq  = (const float*)d_in[2];
  const float* Wk  = (const float*)d_in[3];
  const float* bk  = (const float*)d_in[4];
  const float* Wv  = (const float*)d_in[5];
  const float* bv  = (const float*)d_in[6];
  const float* Wo  = (const float*)d_in[7];
  const float* bo  = (const float*)d_in[8];
  const float* g1  = (const float*)d_in[9];
  const float* be1 = (const float*)d_in[10];
  const float* g2  = (const float*)d_in[11];
  const float* be2 = (const float*)d_in[12];
  const float* W1  = (const float*)d_in[13];
  const float* bf1 = (const float*)d_in[14];
  const float* W2  = (const float*)d_in[15];
  const float* bf2 = (const float*)d_in[16];
  float* out = (float*)d_out;

  constexpr int BS = 16384, D = 256, F = 512;

  // workspace layout (all regions 16B-aligned by construction)
  unsigned short* xn1   = (unsigned short*)d_ws;                 // [BS][D] bf16
  unsigned short* qkvb  = xn1  + (size_t)BS * D;                 // [12][BS][D]
  unsigned short* attnb = qkvb + (size_t)12 * BS * D;            // [4][BS][D]
  float*          x1    = (float*)(attnb + (size_t)4 * BS * D);  // [BS][D] f32
  unsigned short* xn2   = (unsigned short*)(x1 + (size_t)BS * D);
  unsigned short* hid   = xn2 + (size_t)BS * D;                  // [BS][F] bf16
  unsigned short* wqkvT = hid + (size_t)BS * F;                  // [12][D][D]
  unsigned short* woT   = wqkvT + (size_t)12 * D * D;            // [D][4*D]
  unsigned short* w1T   = woT + (size_t)D * 4 * D;               // [F][D]
  unsigned short* w2T   = w1T + (size_t)F * D;                   // [D][F]

  // 0. one-time weight prep: f32 [K][N] -> bf16 [N][K]
  for (int m = 0; m < 12; ++m) {
    const int t = m >> 2, h = m & 3;
    const float* W = (t == 0) ? Wq : (t == 1) ? Wk : Wv;
    wprep_kernel<<<(D * D) / 256, 256, 0, stream>>>(
        W + (size_t)h * D * D, wqkvT + (size_t)m * D * D, D, D);
  }
  wprep_kernel<<<(4 * D * D) / 256, 256, 0, stream>>>(Wo, woT, 4 * D, D);
  wprep_kernel<<<(D * F) / 256, 256, 0, stream>>>(W1, w1T, D, F);
  wprep_kernel<<<(F * D) / 256, 256, 0, stream>>>(W2, w2T, F, D);

  // 1. LN1
  ln_bf16_kernel<<<BS, 256, 0, stream>>>(x, g1, be1, xn1);

  // 2. per-head Q/K/V projections (12 GEMMs, [BS,256]x[256,256])
  for (int t = 0; t < 3; ++t) {
    const float* bb = (t == 0) ? bq : (t == 1) ? bk : bv;
    for (int h = 0; h < 4; ++h) {
      const int m = t * 4 + h;
      gemm_bf16_kernel<true, false, false, false, true>
          <<<dim3(BS / 128, D / 128), 256, 0, stream>>>(
              xn1, D, 8, (size_t)0, wqkvT + (size_t)m * D * D, D, bb + h * D,
              nullptr, nullptr, qkvb + (size_t)m * BS * D, D, D);
    }
  }

  // 3. attention (flash, per batch*head)
  attn_kernel<<<dim3(32, 32), 128, 0, stream>>>(qkvb, attnb);

  // 4. output projection over concatenated heads + residual
  gemm_bf16_kernel<true, true, false, true, false>
      <<<dim3(BS / 128, D / 128), 256, 0, stream>>>(
          attnb, D, 8, (size_t)BS * D, woT, 4 * D, bo,
          x, x1, nullptr, D, 4 * D);

  // 5. LN2
  ln_bf16_kernel<<<BS, 256, 0, stream>>>(x1, g2, be2, xn2);

  // 6. FFN up + ReLU
  gemm_bf16_kernel<true, false, true, false, true>
      <<<dim3(BS / 128, F / 128), 256, 0, stream>>>(
          xn2, D, 8, (size_t)0, w1T, D, bf1,
          nullptr, nullptr, hid, F, D);

  // 7. FFN down + residual -> f32 output
  gemm_bf16_kernel<true, true, false, true, false>
      <<<dim3(BS / 128, D / 128), 256, 0, stream>>>(
          hid, F, 9, (size_t)0, w2T, F, bf2,
          x1, out, nullptr, D, F);
}
